// FeatureSpaceLoss_55422257987622
// MI455X (gfx1250) — matrix operand order repaired and checked by reading, not compiled, not run on hardware
//
#include <hip/hip_runtime.h>
#include <hip/hip_bf16.h>
#include <math.h>

// ---------------------------------------------------------------------------
// FeatureSpaceLoss on MI455X (gfx1250, wave32, WMMA f32_16x16x32_f16)
//
// Math: after row normalization every row norm^2 == D, so
//   softmax(-d^2/tau) == softmax(2*(X@Y^T)/tau)
// and the 3 temperatures collapse into one combined weight matrix
//   U = (1/3) * sum_tau exp(2(S-m)/tau) / l_tau
// so per scale:   S = X Y^T  (WMMA GEMM, f32 out)
//                 U_pos/U_neg from S (rowwise max/denoms, f16 out)
//                 raw = sum_i || (U_pos@Yp - U_neg@Yn)_i ||^2  (WMMA GEMM,
//                       tile squared+reduced in registers, never stored)
// Outputs: loss = dn/D, dn = raw_mean/((sqrt(raw_mean/D)+eps)^2).
//
// GEMM loops are software-pipelined (double-buffered operands) so the
// compiler can issue partial s_wait_loadcnt instead of draining to zero
// before every v_wmma.
// ---------------------------------------------------------------------------

typedef __attribute__((ext_vector_type(16))) _Float16 v16h;
typedef __attribute__((ext_vector_type(8)))  _Float16 v8h;
typedef __attribute__((ext_vector_type(8)))  float    v8f;

#define EPSF   1e-8f
#define NROWS  4096
#define LDF    2048      // feature row stride (f16 elements)
#define LDSQ   4096      // S / U row stride

// ---- WMMA operand loaders (contiguous k order, same for A and B, wave32) ----------
// A (16x32 f16, M x K): lane m=l&15; halves h -> K = (l>=16?16:0)+h.
__device__ __forceinline__ v16h load_a16(const _Float16* __restrict__ base,
                                         int ld, int row0, int k0, int lane) {
    int m  = lane & 15;
    int ks = (lane >> 4) << 4;                 // 0 or 16
    const _Float16* p = base + (size_t)(row0 + m) * (size_t)ld + (size_t)(k0 + ks);
    v8h lo = *(const v8h*)(p);
    v8h hi = *(const v8h*)(p + 8);
    return __builtin_shufflevector(lo, hi, 0,1,2,3,4,5,6,7,8,9,10,11,12,13,14,15);
}

// B (32x16 f16, K x N): lane n=l&15; halves h -> K = (l>=16?16:0)+h.
// B[k][n] taken from row (row0+n) of a row-major matrix, contiguous in k.
__device__ __forceinline__ v16h load_b16(const _Float16* __restrict__ base,
                                         int ld, int row0, int k0, int lane) {
    int n  = lane & 15;
    int ks = (lane >> 4) << 4;                 // 0 or 16
    const _Float16* p = base + (size_t)(row0 + n) * (size_t)ld + (size_t)(k0 + ks);
    return *(const v16h*)(p);
}

__device__ __forceinline__ v8f wmma_f16(v16h a, v16h b, v8f c) {
    return __builtin_amdgcn_wmma_f32_16x16x32_f16(false, a, false, b,
                                                  (short)0, c, false, false);
}

// ---------------------------------------------------------------------------
// Kernel 1: multi-scale feature extraction + per-scale row normalization.
// One block per (array, row). Writes f16 features [4096][2048]; optionally a
// transposed copy [2048][4096] (needed for Y matrices as value-GEMM B operand).
// ---------------------------------------------------------------------------
__global__ __launch_bounds__(256)
void k_extract(const float* __restrict__ A, const float* __restrict__ Wg,
               const float* __restrict__ W2, const float* __restrict__ W4,
               _Float16* __restrict__ feat, _Float16* __restrict__ featT) {
    __shared__ float arow[512];
    __shared__ float amean[32];
    __shared__ float sfeat[2048];
    __shared__ float r0[256], r1[256], r2[256];
    __shared__ float sc[3];
    const int n = blockIdx.x;
    const int t = threadIdx.x;
    const float* a = A + (size_t)n * 512;
    arow[t]       = a[t];
    arow[t + 256] = a[t + 256];
    __syncthreads();
    if (t < 32) {                               // temporal mean for 'global'
        float s = 0.f;
        #pragma unroll
        for (int tt = 0; tt < 16; ++tt) s += arow[tt * 32 + t];
        amean[t] = s * (1.0f / 16.0f);
    }
    __syncthreads();
    float ss0 = 0.f, ss1 = 0.f, ss2 = 0.f;
    #pragma unroll
    for (int it = 0; it < 8; ++it) {
        int c = t + it * 256;
        float v = 0.f;
        if (c < 512) {                          // global: mean @ Wg[32,512]
            for (int d = 0; d < 32; ++d) v += amean[d] * Wg[d * 512 + c];
            ss0 += v * v;
        } else if (c < 1536) {                  // patch_2: [8,64] @ W2[64,128]
            int cc = c - 512, ch = cc >> 7, j = cc & 127;
            const float* ar = arow + ch * 64;
            for (int q = 0; q < 64; ++q) v += ar[q] * W2[q * 128 + j];
            ss1 += v * v;
        } else {                                // patch_4: [4,128] @ W4[128,128]
            int cc = c - 1536, ch = cc >> 7, j = cc & 127;
            const float* ar = arow + ch * 128;
            for (int q = 0; q < 128; ++q) v += ar[q] * W4[q * 128 + j];
            ss2 += v * v;
        }
        sfeat[c] = v;
    }
    r0[t] = ss0; r1[t] = ss1; r2[t] = ss2;
    __syncthreads();
    for (int s = 128; s > 0; s >>= 1) {
        if (t < s) { r0[t] += r0[t+s]; r1[t] += r1[t+s]; r2[t] += r2[t+s]; }
        __syncthreads();
    }
    if (t == 0) {                               // z * sqrt(D)/(||z||+eps)
        sc[0] = sqrtf(512.0f)  / (sqrtf(r0[0]) + EPSF);
        sc[1] = sqrtf(1024.0f) / (sqrtf(r1[0]) + EPSF);
        sc[2] = sqrtf(512.0f)  / (sqrtf(r2[0]) + EPSF);
    }
    __syncthreads();
    #pragma unroll
    for (int it = 0; it < 8; ++it) {
        int c = t + it * 256;
        float s = (c < 512) ? sc[0] : (c < 1536 ? sc[1] : sc[2]);
        _Float16 h = (_Float16)(sfeat[c] * s);
        feat[(size_t)n * LDF + c] = h;
        if (featT) featT[(size_t)c * NROWS + n] = h;
    }
}

// ---------------------------------------------------------------------------
// Kernel 2: S = X @ Y^T  (f16 x f16 -> f32), per scale slice [coff, coff+D).
// Block = 8 waves laid out 4(M) x 2(N); each wave computes a 32x32 tile
// (2 A-tiles x 2 B-tiles -> 4 WMMA chains, 8 b128 loads per K-step).
// Double-buffered over K. Grid: (4096/128, 4096/64).
// ---------------------------------------------------------------------------
__global__ __launch_bounds__(256)
void k_gemm_s(const _Float16* __restrict__ X, const _Float16* __restrict__ Y,
              float* __restrict__ S, int coff, int D) {
    const int lane = threadIdx.x & 31;
    const int wave = threadIdx.x >> 5;
    const int i0 = blockIdx.x * 128 + (wave & 3) * 32;
    const int j0 = blockIdx.y * 64  + (wave >> 2) * 32;
    v8f acc00 = {}, acc01 = {}, acc10 = {}, acc11 = {};
    // prologue: load K-step 0
    v16h a0 = load_a16(X, LDF, i0,      coff, lane);
    v16h a1 = load_a16(X, LDF, i0 + 16, coff, lane);
    v16h b0 = load_b16(Y, LDF, j0,      coff, lane);
    v16h b1 = load_b16(Y, LDF, j0 + 16, coff, lane);
    for (int k0 = 0; k0 < D; k0 += 32) {
        // prefetch next K-step (clamped to 0 on last iter; always valid mem)
        const int kn = (k0 + 32 < D) ? (k0 + 32) : 0;
        v16h a0n = load_a16(X, LDF, i0,      coff + kn, lane);
        v16h a1n = load_a16(X, LDF, i0 + 16, coff + kn, lane);
        v16h b0n = load_b16(Y, LDF, j0,      coff + kn, lane);
        v16h b1n = load_b16(Y, LDF, j0 + 16, coff + kn, lane);
        acc00 = wmma_f16(a0, b0, acc00);
        acc01 = wmma_f16(a0, b1, acc01);
        acc10 = wmma_f16(a1, b0, acc10);
        acc11 = wmma_f16(a1, b1, acc11);
        a0 = a0n; a1 = a1n; b0 = b0n; b1 = b1n;
    }
    // C/D layout: VGPR r, lanes 0-15 -> M=r, lanes 16-31 -> M=r+8; N=lane&15.
    const int mo = (lane >> 4) << 3;
    const int nn = lane & 15;
    #pragma unroll
    for (int r = 0; r < 8; ++r) {
        size_t row0 = (size_t)(i0 + mo + r) * LDSQ;
        size_t row1 = (size_t)(i0 + 16 + mo + r) * LDSQ;
        S[row0 + j0      + nn] = acc00[r];
        S[row0 + j0 + 16 + nn] = acc01[r];
        S[row1 + j0      + nn] = acc10[r];
        S[row1 + j0 + 16 + nn] = acc11[r];
    }
}

// ---------------------------------------------------------------------------
// Kernel 3: per-row softmax stats + combined-temperature weight matrix.
// 2/tau for tau = 0.02, 0.05, 0.2  ->  100, 40, 10.
// Block per row; three passes over a 16KB row (L2/WGP$-resident).
// ---------------------------------------------------------------------------
__global__ __launch_bounds__(256)
void k_softmax_u(const float* __restrict__ S, _Float16* __restrict__ U) {
    __shared__ float r0[256], r1[256], r2[256];
    __shared__ float stat[4];
    const int i = blockIdx.x, t = threadIdx.x;
    const float* row = S + (size_t)i * LDSQ;
    float mx = -3.0e38f;
    for (int j = t; j < LDSQ; j += 256) mx = fmaxf(mx, row[j]);
    r0[t] = mx; __syncthreads();
    for (int s = 128; s > 0; s >>= 1) {
        if (t < s) r0[t] = fmaxf(r0[t], r0[t + s]);
        __syncthreads();
    }
    if (t == 0) stat[0] = r0[0];
    __syncthreads();
    const float m = stat[0];
    float e0 = 0.f, e1 = 0.f, e2 = 0.f;
    for (int j = t; j < LDSQ; j += 256) {
        float d = row[j] - m;
        e0 += __expf(100.f * d);
        e1 += __expf(40.f  * d);
        e2 += __expf(10.f  * d);
    }
    r0[t] = e0; r1[t] = e1; r2[t] = e2;
    __syncthreads();
    for (int s = 128; s > 0; s >>= 1) {
        if (t < s) { r0[t] += r0[t+s]; r1[t] += r1[t+s]; r2[t] += r2[t+s]; }
        __syncthreads();
    }
    if (t == 0) { stat[1] = 1.f / r0[0]; stat[2] = 1.f / r1[0]; stat[3] = 1.f / r2[0]; }
    __syncthreads();
    const float i0 = stat[1], i1 = stat[2], i2 = stat[3];
    _Float16* urow = U + (size_t)i * LDSQ;
    for (int j = t; j < LDSQ; j += 256) {
        float d = row[j] - m;
        float u = (1.0f / 3.0f) *
                  (__expf(100.f * d) * i0 + __expf(40.f * d) * i1 + __expf(10.f * d) * i2);
        urow[j] = (_Float16)u;
    }
}

// ---------------------------------------------------------------------------
// Kernel 4: raw[s] partials = sum over V-tile of (Upos@Yp - Uneg@Yn)^2.
// Wave computes a 16x32 V-tile fully reduced over K=4096 in registers
// (4 WMMA accumulator chains: pos/neg x 2 N-tiles), double-buffered over K,
// squares, block-reduces, writes one deterministic partial per block.
// Grid: (4096/128, D/32).
// ---------------------------------------------------------------------------
__global__ __launch_bounds__(256)
void k_vnorm(const _Float16* __restrict__ Up, const _Float16* __restrict__ Un,
             const _Float16* __restrict__ YTp, const _Float16* __restrict__ YTn,
             int coff, float* __restrict__ part) {
    __shared__ float red[256];
    const int lane = threadIdx.x & 31;
    const int wave = threadIdx.x >> 5;
    const int i0 = blockIdx.x * 128 + wave * 16;
    const int c0 = coff + blockIdx.y * 32;     // feature-column = YT row
    v8f ap0 = {}, ap1 = {}, an0 = {}, an1 = {};
    // prologue
    v16h uap = load_a16(Up, LDSQ, i0, 0, lane);
    v16h uan = load_a16(Un, LDSQ, i0, 0, lane);
    v16h bp0 = load_b16(YTp, NROWS, c0,      0, lane);
    v16h bp1 = load_b16(YTp, NROWS, c0 + 16, 0, lane);
    v16h bn0 = load_b16(YTn, NROWS, c0,      0, lane);
    v16h bn1 = load_b16(YTn, NROWS, c0 + 16, 0, lane);
    for (int k0 = 0; k0 < NROWS; k0 += 32) {
        const int kn = (k0 + 32 < NROWS) ? (k0 + 32) : 0;
        v16h uapn = load_a16(Up, LDSQ, i0, kn, lane);
        v16h uann = load_a16(Un, LDSQ, i0, kn, lane);
        v16h bp0n = load_b16(YTp, NROWS, c0,      kn, lane);
        v16h bp1n = load_b16(YTp, NROWS, c0 + 16, kn, lane);
        v16h bn0n = load_b16(YTn, NROWS, c0,      kn, lane);
        v16h bn1n = load_b16(YTn, NROWS, c0 + 16, kn, lane);
        ap0 = wmma_f16(uap, bp0, ap0);
        ap1 = wmma_f16(uap, bp1, ap1);
        an0 = wmma_f16(uan, bn0, an0);
        an1 = wmma_f16(uan, bn1, an1);
        uap = uapn; uan = uann;
        bp0 = bp0n; bp1 = bp1n; bn0 = bn0n; bn1 = bn1n;
    }
    float ssq = 0.f;
    #pragma unroll
    for (int r = 0; r < 8; ++r) {
        float v0 = ap0[r] - an0[r];
        float v1 = ap1[r] - an1[r];
        ssq += v0 * v0 + v1 * v1;
    }
    red[threadIdx.x] = ssq; __syncthreads();
    for (int s = 128; s > 0; s >>= 1) {
        if (threadIdx.x < s) red[threadIdx.x] += red[threadIdx.x + s];
        __syncthreads();
    }
    if (threadIdx.x == 0) part[blockIdx.x * gridDim.y + blockIdx.y] = red[0];
}

// ---------------------------------------------------------------------------
// Kernel 5: fold per-block partials into the two output scalars.
// raw_mean = raw/N; lambda = sqrt(raw_mean/D); dn = raw_mean/(lambda+eps)^2;
// loss = dn/D.
// ---------------------------------------------------------------------------
__global__ __launch_bounds__(256)
void k_final(const float* __restrict__ part, float* __restrict__ out) {
    __shared__ float red[256];
    const int t = threadIdx.x;
    float tl = 0.f, td = 0.f;
    const int Ds[3] = {512, 1024, 512};
    for (int s = 0; s < 3; ++s) {
        float acc = 0.f;
        for (int j = t; j < 1024; j += 256) acc += part[s * 1024 + j];
        red[t] = acc; __syncthreads();
        for (int k = 128; k > 0; k >>= 1) {
            if (t < k) red[t] += red[t + k];
            __syncthreads();
        }
        if (t == 0) {
            float rn  = red[0] / 4096.0f;
            float D   = (float)Ds[s];
            float lam = sqrtf(rn / D);
            float dn  = rn / ((lam + EPSF) * (lam + EPSF));
            tl += dn / D;
            td += dn;
        }
        __syncthreads();
    }
    if (t == 0) { out[0] = tl; out[1] = td * (1.0f / 3.0f); }
}

__global__ void k_zero(float* __restrict__ p, int n) {
    int i = blockIdx.x * 256 + threadIdx.x;
    if (i < n) p[i] = 0.f;
}

// ---------------------------------------------------------------------------
extern "C" void kernel_launch(void* const* d_in, const int* in_sizes, int n_in,
                              void* d_out, int out_size, void* d_ws, size_t ws_size,
                              hipStream_t stream) {
    (void)in_sizes; (void)n_in; (void)out_size; (void)ws_size;
    const float* actions = (const float*)d_in[0];
    const float* apos    = (const float*)d_in[1];
    const float* aneg    = (const float*)d_in[2];
    const float* Wg      = (const float*)d_in[3];
    const float* W2      = (const float*)d_in[4];
    const float* W4      = (const float*)d_in[5];
    float* out = (float*)d_out;

    // Workspace layout (~208 MB):
    const size_t FE = (size_t)NROWS * LDF;        // 8M f16 per feature matrix
    const size_t SE = (size_t)NROWS * LDSQ;       // 16M f32 / f16 per square mat
    _Float16* fX  = (_Float16*)d_ws;
    _Float16* fP  = fX  + FE;
    _Float16* fN  = fP  + FE;
    _Float16* fTP = fN  + FE;                     // transposed pos features
    _Float16* fTN = fTP + FE;                     // transposed neg features
    float*    S   = (float*)(fTN + FE);           // 64 MB logits scratch
    _Float16* Up  = (_Float16*)(S + SE);          // 32 MB combined weights (pos)
    _Float16* Un  = Up + SE;                      // 32 MB combined weights (neg)
    float*    part = (float*)(Un + SE);           // 3 * 1024 partials

    k_zero<<<12, 256, 0, stream>>>(part, 3 * 1024);

    k_extract<<<NROWS, 256, 0, stream>>>(actions, Wg, W2, W4, fX, nullptr);
    k_extract<<<NROWS, 256, 0, stream>>>(apos,    Wg, W2, W4, fP, fTP);
    k_extract<<<NROWS, 256, 0, stream>>>(aneg,    Wg, W2, W4, fN, fTN);

    const int coffs[3] = {0, 512, 1536};
    const int Ds[3]    = {512, 1024, 512};
    for (int s = 0; s < 3; ++s) {
        const int coff = coffs[s], D = Ds[s];
        // positives
        k_gemm_s<<<dim3(NROWS / 128, NROWS / 64), 256, 0, stream>>>(fX, fP, S, coff, D);
        k_softmax_u<<<NROWS, 256, 0, stream>>>(S, Up);
        // negatives (S buffer reused)
        k_gemm_s<<<dim3(NROWS / 128, NROWS / 64), 256, 0, stream>>>(fX, fN, S, coff, D);
        k_softmax_u<<<NROWS, 256, 0, stream>>>(S, Un);
        // raw drift-norm partials for this scale
        k_vnorm<<<dim3(NROWS / 128, D / 32), 256, 0, stream>>>(
            Up, Un, fTP, fTN, coff, part + s * 1024);
    }

    k_final<<<1, 256, 0, stream>>>(part, out);
}
